// DCGRUClassifier_4037269258969
// MI455X (gfx1250) — matrix operand order repaired and checked by reading, not compiled, not run on hardware
//
#include <hip/hip_runtime.h>

// ---------------------------------------------------------------------------
// DCGRU classifier for MI455X (gfx1250).
// B=32 T=256 N=128 DIN=16 U=64 K=2 C=4, M=3 diffusion matrices.
// One persistent workgroup (8 wave32) per batch; activations LDS-resident in
// channel-major layout; support-matrix A-fragments pinned in registers for the
// whole kernel; all matmuls via v_wmma_f32_16x16x32_f16 (f32 accumulate).
// Wave w owns node-row-block w (mt = wave) in every matmul. Projections use
// kt-outer / nt-inner order with an accumulator array: only 2 A fragments
// live at a time (vs KT), B loads per kt are independent (progressive waits),
// and consecutive WMMAs hit different accumulators -> no same-register D->C
// chains, no scratch spills.
// ---------------------------------------------------------------------------

typedef __attribute__((ext_vector_type(16))) _Float16 v16h;
typedef __attribute__((ext_vector_type(8)))  _Float16 v8h;
typedef __attribute__((ext_vector_type(8)))  float    v8f;
typedef __attribute__((ext_vector_type(4)))  float    v4f;

#define WMMA_F16(a, b, c) \
  __builtin_amdgcn_wmma_f32_16x16x32_f16(false, (a), false, (b), (short)0, (c), false, false)

namespace {

constexpr int Bn  = 32;    // batch
constexpr int Tn  = 256;   // timesteps
constexpr int Nn  = 128;   // graph nodes
constexpr int DIN = 16;
constexpr int Un  = 64;    // hidden units
constexpr int Cn  = 4;     // classes
constexpr int LD  = 136;   // LDS leading dim (halves): rows 16B aligned, banks spread

// ws layout (in _Float16 elements): transposed (Nout x Kpad) f16 weights with
// rows permuted from (c*3+m) interleave to [X0 | X1 | X2] block order.
constexpr int WG0T_OFF = 0;                          // 128 x 256
constexpr int WC0T_OFF = WG0T_OFF + 128 * 256;       //  64 x 256
constexpr int WG1T_OFF = WC0T_OFF + 64  * 256;       // 128 x 384
constexpr int WC1T_OFF = WG1T_OFF + 128 * 384;       //  64 x 384
constexpr int WS_HALVES = WC1T_OFF + 64 * 384;       // 122880 halves = 240KB

} // namespace

// --- fragment helpers (ISA 7.12.2 layouts, wave32) -------------------------

// A fragment where A[m][k] = matT[k][m] (channel-major activations).
__device__ __forceinline__ v16h load_a_kmajor(const _Float16* matT, int ld,
                                              int m0, int k0, int lane) {
  int m  = m0 + (lane & 15);
  int kb = k0 + ((lane >> 4) << 3);
  v16h a;
#pragma unroll
  for (int h = 0; h < 16; ++h) {
    int kk = kb + h + ((h & 8) ? 8 : 0);   // h<8: kb+h ; h>=8: kb+16+(h-8)
    a[h] = matT[kk * ld + m];
  }
  return a;
}

// B fragment (32x16 f16) where B[k][n] = matT[n][k] (rows = output columns,
// contiguous K). Lanes 0..15: n=lane, K=k0..7,16..23; lanes 16..31: K=8..15,24..31.
__device__ __forceinline__ v16h load_b_nmajor(const _Float16* matT, int ld,
                                              int k0, int n0, int lane) {
  int n  = n0 + (lane & 15);
  int kb = k0 + ((lane >> 4) << 3);
  const _Float16* p = matT + n * ld + kb;
  v8h lo = *(const v8h*)p;
  v8h hi = *(const v8h*)(p + 16);
  v16h b;
#pragma unroll
  for (int i = 0; i < 8; ++i) { b[i] = lo[i]; b[i + 8] = hi[i]; }
  return b;
}

// Shared projection core: acc[nt] += Xc-rowblock(wave) @ WT-coltile(nt).
// kt-outer / nt-inner; A fragment software-pipelined one kt ahead.
template <int KT, int NT>
__device__ __forceinline__ void proj_core(const _Float16* sXT,
                                          const _Float16* WT,
                                          v8f (&acc)[NT], int wave, int lane) {
  v16h af = load_a_kmajor(sXT, LD, wave * 16, 0, lane);
  for (int kt = 0; kt < KT; ++kt) {
    v16h afn;
    if (kt + 1 < KT)
      afn = load_a_kmajor(sXT, LD, wave * 16, (kt + 1) * 32, lane);
#pragma unroll
    for (int nt = 0; nt < NT; ++nt) {
      v16h bm = load_b_nmajor(WT, KT * 32, kt * 32, nt * 16, lane);
      acc[nt] = WMMA_F16(af, bm, acc[nt]);
    }
    if (kt + 1 < KT) af = afn;
  }
}

// Gate projection: sGT = sigmoid(Xc @ Wg + bg), 128 output channels.
template <int KT>
__device__ __forceinline__ void proj_gate(const _Float16* sXT, _Float16* sGT,
                                          const _Float16* WT,
                                          const float* __restrict__ bias,
                                          int wave, int lane) {
  v8f acc[8] = {};
  proj_core<KT, 8>(sXT, WT, acc, wave, lane);
  const int mb = wave * 16 + ((lane >> 4) << 3);
#pragma unroll
  for (int nt = 0; nt < 8; ++nt) {
    int ch = nt * 16 + (lane & 15);
    float bs = bias[ch];
    v8h dh;
#pragma unroll
    for (int j = 0; j < 8; ++j)
      dh[j] = (_Float16)(1.0f / (1.0f + __expf(-(acc[nt][j] + bs))));
    *(v8h*)(sGT + ch * LD + mb) = dh;
  }
}

// Candidate projection + GRU update: h = u*h + (1-u)*tanh(Xc@Wc + bc).
template <int KT>
__device__ __forceinline__ void proj_cand(const _Float16* sXT,
                                          const _Float16* sGT, _Float16* sH,
                                          const _Float16* WT,
                                          const float* __restrict__ bias,
                                          int wave, int lane) {
  v8f acc[4] = {};
  proj_core<KT, 4>(sXT, WT, acc, wave, lane);
  const int mb = wave * 16 + ((lane >> 4) << 3);
#pragma unroll
  for (int nt = 0; nt < 4; ++nt) {
    int ch = nt * 16 + (lane & 15);
    float bs = bias[ch];
    v8h uh = *(const v8h*)(sGT + (Un + ch) * LD + mb);
    v8h hh = *(const v8h*)(sH + ch * LD + mb);
    v8h dh;
#pragma unroll
    for (int j = 0; j < 8; ++j) {
      float cnd = tanhf(acc[nt][j] + bs);
      float u   = (float)uh[j];
      dh[j] = (_Float16)(u * (float)hh[j] + (1.0f - u) * cnd);
    }
    *(v8h*)(sH + ch * LD + mb) = dh;
  }
}

// --- weight prep: f32 -> f16, row-permute (c*3+m -> m*Cin+c), transpose, pad
__global__ void dcgru_prep_weights(const float* __restrict__ Wg0,
                                   const float* __restrict__ Wc0,
                                   const float* __restrict__ Wg1,
                                   const float* __restrict__ Wc1,
                                   _Float16* __restrict__ ws) {
  int i = blockIdx.x * blockDim.x + threadIdx.x;
  if (i >= WS_HALVES) return;
  float v = 0.0f;
  if (i < WC0T_OFF) {                       // Wg0T: 128 x 256 (K real = 240)
    int j = i - WG0T_OFF, n = j >> 8, k = j & 255;
    if (k < 240) { int c = k % 80, mm = k / 80; v = Wg0[(c * 3 + mm) * 128 + n]; }
  } else if (i < WG1T_OFF) {                // Wc0T: 64 x 256
    int j = i - WC0T_OFF, n = j >> 8, k = j & 255;
    if (k < 240) { int c = k % 80, mm = k / 80; v = Wc0[(c * 3 + mm) * 64 + n]; }
  } else if (i < WC1T_OFF) {                // Wg1T: 128 x 384
    int j = i - WG1T_OFF, n = j / 384, k = j % 384;
    int c = k % 128, mm = k / 128;
    v = Wg1[(c * 3 + mm) * 128 + n];
  } else {                                  // Wc1T: 64 x 384
    int j = i - WC1T_OFF, n = j / 384, k = j % 384;
    int c = k % 128, mm = k / 128;
    v = Wc1[(c * 3 + mm) * 64 + n];
  }
  ws[i] = (_Float16)v;
}

// --- persistent recurrent kernel -------------------------------------------

__global__ __launch_bounds__(256) void dcgru_main(
    const float* __restrict__ xseq,      // (B,T,N,DIN)
    const int*   __restrict__ slen,      // (B,)
    const float* __restrict__ support,   // (N,N)
    const _Float16* __restrict__ Wg0T, const _Float16* __restrict__ Wc0T,
    const _Float16* __restrict__ Wg1T, const _Float16* __restrict__ Wc1T,
    const float* __restrict__ bg0, const float* __restrict__ bc0,
    const float* __restrict__ bg1, const float* __restrict__ bc1,
    const float* __restrict__ Wfc, const float* __restrict__ bfc,
    float* __restrict__ out) {
  // LDS (all activations channel-major: row = channel, col = node)
  __shared__ _Float16 sXT[384 * LD];   // [X0 | X1 | X2] diffusion workspace
  __shared__ _Float16 sGT[128 * LD];   // gate outputs (r: rows 0..63, u: 64..127)
  __shared__ _Float16 sH1[Un * LD];    // layer-0 hidden
  __shared__ _Float16 sH2[Un * LD];    // layer-1 hidden
  __shared__ float    sRed[Nn * Cn];   // output reduction

  const int b    = blockIdx.x;
  const int tid  = threadIdx.x;
  const int wave = tid >> 5;
  const int lane = tid & 31;
  const int nlo  = lane & 15;
  const int mb   = wave * 16 + ((lane >> 4) << 3);  // D-store column base

  // Support A-fragments: wave w holds node-row-block w of S for the entire
  // kernel (support is time-invariant). Loaded straight from global f32.
  v16h sA[4];
  {
    int m = wave * 16 + nlo;
#pragma unroll
    for (int kt = 0; kt < 4; ++kt) {
      int kb = kt * 32 + ((lane >> 4) << 3);
      v16h a;
#pragma unroll
      for (int h = 0; h < 16; ++h) {
        int kk = kb + h + ((h & 8) ? 8 : 0);
        a[h] = (_Float16)support[m * Nn + kk];
      }
      sA[kt] = a;
    }
  }

  // zero hidden states (vectorized)
  {
    v8h z = {};
    for (int e = tid; e < Un * LD / 8; e += 256) {
      *(v8h*)(sH1 + e * 8) = z;
      *(v8h*)(sH2 + e * 8) = z;
    }
  }
  const int last_t = slen[b] - 1;
  __syncthreads();

  // Diffusion: Xdst = S @ Xsrc (or 2*S@Xsrc - X0 for the Chebyshev term).
  // mt = wave; support A-fragments live in registers; B double-buffered.
  auto diffusion = [&](int cin_tiles, int src_row, int dst_row, bool cheb) {
    const _Float16* src = sXT + src_row * LD;
    v16h bf[4];
#pragma unroll
    for (int kt = 0; kt < 4; ++kt)
      bf[kt] = load_b_nmajor(src, LD, kt * 32, 0, lane);
    for (int ct = 0; ct < cin_tiles; ++ct) {
      v16h bn[4];
      if (ct + 1 < cin_tiles) {
#pragma unroll
        for (int kt = 0; kt < 4; ++kt)
          bn[kt] = load_b_nmajor(src, LD, kt * 32, (ct + 1) * 16, lane);
      }
      v8f acc = {};
#pragma unroll
      for (int kt = 0; kt < 4; ++kt) acc = WMMA_F16(sA[kt], bf[kt], acc);
      int c = ct * 16 + nlo;
      if (cheb) {
        v8h x0 = *(const v8h*)(sXT + c * LD + mb);   // X0 block at row 0
#pragma unroll
        for (int j = 0; j < 8; ++j) acc[j] = 2.0f * acc[j] - (float)x0[j];
      }
      v8h dh;
#pragma unroll
      for (int j = 0; j < 8; ++j) dh[j] = (_Float16)acc[j];
      *(v8h*)(sXT + (dst_row + c) * LD + mb) = dh;
      if (ct + 1 < cin_tiles) {
#pragma unroll
        for (int kt = 0; kt < 4; ++kt) bf[kt] = bn[kt];
      }
    }
  };

  for (int t = 0; t < Tn; ++t) {
    // ================= layer 0 (Cin = 16+64 = 80, Kpad = 256) ==============
    const float* xp = xseq + ((size_t)(b * Tn + t)) * (size_t)(Nn * DIN);
    if (t + 1 < Tn)                                  // prefetch next x_t slab
      __builtin_prefetch(xp + Nn * DIN + tid * 32, 0, 1);
    for (int e = tid; e < (DIN * Nn) / 4; e += 256) { // x_t -> rows 0..15
      int n = e >> 2, cg = e & 3;
      v4f v = ((const v4f*)xp)[n * 4 + cg];
#pragma unroll
      for (int j = 0; j < 4; ++j)
        sXT[(cg * 4 + j) * LD + n] = (_Float16)v[j];
    }
    for (int e = tid; e < Un * (Nn / 8); e += 256) { // h1 -> rows 16..79
      int c = e >> 4, q = (e & 15) * 8;
      *(v8h*)(sXT + (DIN + c) * LD + q) = *(const v8h*)(sH1 + c * LD + q);
    }
    {
      v8h z = {};
      for (int e = tid; e < 272; e += 256)           // zero pad rows 240..255
        *(v8h*)(sXT + 240 * LD + e * 8) = z;
    }
    __syncthreads();

    diffusion(5, 0, 80, false);  __syncthreads();    // X1 = S X0
    diffusion(5, 80, 160, true); __syncthreads();    // X2 = 2 S X1 - X0
    proj_gate<8>(sXT, sGT, Wg0T, bg0, wave, lane); __syncthreads();

    for (int e = tid; e < Un * (Nn / 8); e += 256) { // rows 16..79 = r * h1
      int c = e >> 4, q = (e & 15) * 8;
      v8h r = *(const v8h*)(sGT + c * LD + q);
      v8h h = *(const v8h*)(sH1 + c * LD + q);
      v8h p;
#pragma unroll
      for (int j = 0; j < 8; ++j)
        p[j] = (_Float16)((float)r[j] * (float)h[j]);
      *(v8h*)(sXT + (DIN + c) * LD + q) = p;
    }
    __syncthreads();
    diffusion(5, 0, 80, false);  __syncthreads();
    diffusion(5, 80, 160, true); __syncthreads();
    proj_cand<8>(sXT, sGT, sH1, Wc0T, bc0, wave, lane); __syncthreads();

    // ================= layer 1 (Cin = 64+64 = 128, K = 384) ================
    for (int e = tid; e < Un * (Nn / 8); e += 256) { // rows 0..63 = h1
      int c = e >> 4, q = (e & 15) * 8;              // rows 64..127 = h2
      *(v8h*)(sXT + c * LD + q)        = *(const v8h*)(sH1 + c * LD + q);
      *(v8h*)(sXT + (Un + c) * LD + q) = *(const v8h*)(sH2 + c * LD + q);
    }
    __syncthreads();

    diffusion(8, 0, 128, false);  __syncthreads();
    diffusion(8, 128, 256, true); __syncthreads();
    proj_gate<12>(sXT, sGT, Wg1T, bg1, wave, lane); __syncthreads();

    for (int e = tid; e < Un * (Nn / 8); e += 256) { // rows 64..127 = r * h2
      int c = e >> 4, q = (e & 15) * 8;
      v8h r = *(const v8h*)(sGT + c * LD + q);
      v8h h = *(const v8h*)(sH2 + c * LD + q);
      v8h p;
#pragma unroll
      for (int j = 0; j < 8; ++j)
        p[j] = (_Float16)((float)r[j] * (float)h[j]);
      *(v8h*)(sXT + (Un + c) * LD + q) = p;
    }
    __syncthreads();
    diffusion(8, 0, 128, false);  __syncthreads();
    diffusion(8, 128, 256, true); __syncthreads();
    proj_cand<12>(sXT, sGT, sH2, Wc1T, bc1, wave, lane); __syncthreads();

    // ================= readout at t = seq_len-1 ============================
    if (t == last_t) {
      if (tid < Nn) {
        float acc[Cn] = {bfc[0], bfc[1], bfc[2], bfc[3]};
        for (int ch = 0; ch < Un; ++ch) {
          float v = (float)sH2[ch * LD + tid];
          v = v > 0.0f ? v : 0.0f;                   // relu
#pragma unroll
          for (int c = 0; c < Cn; ++c) acc[c] += v * Wfc[ch * Cn + c];
        }
#pragma unroll
        for (int c = 0; c < Cn; ++c) sRed[tid * Cn + c] = acc[c];
      }
      __syncthreads();
      if (tid < Cn) {
        float mx = sRed[tid];
        for (int n = 1; n < Nn; ++n) mx = fmaxf(mx, sRed[n * Cn + tid]);
        out[b * Cn + tid] = mx;                      // max-pool over nodes
      }
      __syncthreads();
    }
  }
}

// ---------------------------------------------------------------------------

extern "C" void kernel_launch(void* const* d_in, const int* in_sizes, int n_in,
                              void* d_out, int out_size, void* d_ws, size_t ws_size,
                              hipStream_t stream) {
  (void)in_sizes; (void)n_in; (void)out_size; (void)ws_size;
  const float* xseq    = (const float*)d_in[0];
  const int*   slen    = (const int*)d_in[1];
  const float* support = (const float*)d_in[2];
  const float* Wg0     = (const float*)d_in[3];
  const float* bg0     = (const float*)d_in[4];
  const float* Wc0     = (const float*)d_in[5];
  const float* bc0     = (const float*)d_in[6];
  const float* Wg1     = (const float*)d_in[7];
  const float* bg1     = (const float*)d_in[8];
  const float* Wc1     = (const float*)d_in[9];
  const float* bc1     = (const float*)d_in[10];
  const float* Wfc     = (const float*)d_in[11];
  const float* bfc     = (const float*)d_in[12];
  float* out = (float*)d_out;
  _Float16* ws = (_Float16*)d_ws;   // needs 240KB; harness scratch

  dcgru_prep_weights<<<(WS_HALVES + 255) / 256, 256, 0, stream>>>(
      Wg0, Wc0, Wg1, Wc1, ws);

  dcgru_main<<<Bn, 256, 0, stream>>>(
      xseq, slen, support,
      ws + WG0T_OFF, ws + WC0T_OFF, ws + WG1T_OFF, ws + WC1T_OFF,
      bg0, bc0, bg1, bc1, Wfc, bfc, out);
}
